// Qwen3MoELayer_45595372814859
// MI455X (gfx1250) — compile-verified, matrix operand order checked by reading, not run on hardware
//
#include <hip/hip_runtime.h>
#include <hip/hip_bf16.h>

typedef __bf16 bf16;
typedef __attribute__((ext_vector_type(16))) __bf16 v16bf;
typedef __attribute__((ext_vector_type(8)))  float  v8f;
typedef __attribute__((ext_vector_type(4)))  int    v4i;

// ---------------- problem constants ----------------
constexpr int BB = 2, NN = 2048, DD = 1024;
constexpr int HH = 16, HD = 64;
constexpr int EE = 8, FF = 4096;
constexpr int MT = BB * NN;            // 4096 tokens

// ---------------- gfx1250 async global->LDS path (guarded) ----------------
#if defined(__HIP_DEVICE_COMPILE__) && \
    __has_builtin(__builtin_amdgcn_global_load_async_to_lds_b128) && \
    __has_builtin(__builtin_amdgcn_s_wait_asynccnt)
#define USE_ASYNC_LDS 1
#else
#define USE_ASYNC_LDS 0
#endif

#if USE_ASYNC_LDS
typedef __attribute__((address_space(1))) v4i* gv4i_p;
typedef __attribute__((address_space(3))) v4i* lv4i_p;
#endif

__device__ inline void async_cp16(const bf16* g, bf16* l) {
#if USE_ASYNC_LDS
  __builtin_amdgcn_global_load_async_to_lds_b128(
      (gv4i_p)(v4i*)g, (lv4i_p)(v4i*)l, 0, 0);
#else
  *(uint4*)l = *(const uint4*)g;
#endif
}

template <int N>
__device__ inline void wait_async() {
#if USE_ASYNC_LDS
  __builtin_amdgcn_s_wait_asynccnt(N);
#endif
}

// ---------------- WMMA helpers ----------------
__device__ inline v8f wmma_bf16(v16bf a, v16bf b, v8f c) {
  return __builtin_amdgcn_wmma_f32_16x16x32_bf16(
      /*neg_a=*/false, a, /*neg_b=*/false, b,
      /*c_mod=*/(short)0, c, /*reuse_a=*/false, /*reuse_b=*/false);
}

// Load one 16x32 bf16 fragment from an LDS tile stored row-major [row][k], ld elems.
// 16-bit A/B layout: lanes 0-15 hold k = kb+{0..7,16..23},
// lanes 16-31 hold k = kb+{8..15,24..31}; row = rowBase + (lane&15).
__device__ inline v16bf load_frag(const bf16* s, int rowBase, int kb, int ld, int lane) {
  const bf16* p = s + (size_t)(rowBase + (lane & 15)) * ld + kb + ((lane >> 4) << 3);
  v16bf f;
  __builtin_memcpy(&f, p, 16);
  __builtin_memcpy(reinterpret_cast<char*>(&f) + 16, p + 16, 16);
  return f;
}

// ---------------- elementwise fp32 -> bf16 ----------------
__global__ __launch_bounds__(256) void k_cvt_bf16(const float* __restrict__ in,
                                                  bf16* __restrict__ out, long n) {
  long i = (long)blockIdx.x * 256 + threadIdx.x;
  long stride = (long)gridDim.x * 256;
  for (; i < n; i += stride) out[i] = (bf16)in[i];
}

// ---------------- RMSNorm (fp32 in, bf16 out) ----------------
__global__ __launch_bounds__(256) void k_rmsnorm(const float* __restrict__ x,
                                                 const float* __restrict__ w,
                                                 bf16* __restrict__ out) {
  __shared__ float red[256];
  int m = blockIdx.x, t = threadIdx.x;
  const float* xr = x + (size_t)m * DD;
  float v[4], s = 0.f;
#pragma unroll
  for (int i = 0; i < 4; ++i) { v[i] = xr[t + i * 256]; s += v[i] * v[i]; }
  red[t] = s; __syncthreads();
  for (int o = 128; o; o >>= 1) { if (t < o) red[t] += red[t + o]; __syncthreads(); }
  float inv = rsqrtf(red[0] / (float)DD + 1e-6f);
#pragma unroll
  for (int i = 0; i < 4; ++i)
    out[(size_t)m * DD + t + i * 256] = (bf16)(v[i] * inv * w[t + i * 256]);
}

// ---------------- tiled WMMA GEMM: C[M,N] = A[M,K] * W[N,K]^T ----------------
// Double-buffered LDS tiles, async global->LDS copies overlapped with WMMA.
// EPI: 0 = store bf16, 1 = store f32, 2 = SiLU -> bf16,
//      3 = out = resid + C * sigmoid(gpre)  (f32),
//      4 = out += coef[row*cs] * C          (f32 accumulate)
enum { EPI_BF16 = 0, EPI_F32 = 1, EPI_SILU = 2, EPI_ATTN = 3, EPI_ACC = 4 };

template <int EPI>
__global__ __launch_bounds__(256)
void k_gemm(const bf16* __restrict__ A, const bf16* __restrict__ W,
            int Ndim, int Kdim,
            float* __restrict__ outF, bf16* __restrict__ outB,
            const float* __restrict__ gpre, const float* __restrict__ resid,
            const float* __restrict__ coef, int cs) {
  __shared__ __align__(16) bf16 sA[2][128 * 32];
  __shared__ __align__(16) bf16 sB[2][128 * 32];
  int t = threadIdx.x, lane = t & 31, wave = t >> 5;
  int bm = blockIdx.y * 128, bn = blockIdx.x * 128;
  int wm = (wave >> 2) * 64;            // 2 waves in M
  int wn = (wave & 3) * 32;             // 4 waves in N
  v8f acc[4][2] = {};
  int r = t >> 2, c8 = (t & 3) * 8;     // cooperative tile loads

  auto issue = [&](int buf, int k0) {
    const bf16* ga = A + (size_t)(bm + r) * Kdim + k0 + c8;
    const bf16* gw = W + (size_t)(bn + r) * Kdim + k0 + c8;
    async_cp16(ga,                      &sA[buf][r * 32 + c8]);
    async_cp16(ga + (size_t)64 * Kdim,  &sA[buf][(r + 64) * 32 + c8]);
    async_cp16(gw,                      &sB[buf][r * 32 + c8]);
    async_cp16(gw + (size_t)64 * Kdim,  &sB[buf][(r + 64) * 32 + c8]);
  };

  int nk = Kdim >> 5;
  issue(0, 0);
  for (int kk = 0; kk < nk; ++kk) {
    int cur = kk & 1;
    if (kk + 1 < nk) {
      issue(cur ^ 1, (kk + 1) << 5);    // prefetch next tile into other buffer
      wait_async<4>();                  // in-order: first 4 (current tile) done
    } else {
      wait_async<0>();
    }
    __syncthreads();
    v16bf af[4], bw[2];
#pragma unroll
    for (int i = 0; i < 4; ++i) af[i] = load_frag(sA[cur], wm + i * 16, 0, 32, lane);
#pragma unroll
    for (int j = 0; j < 2; ++j) bw[j] = load_frag(sB[cur], wn + j * 16, 0, 32, lane);
#pragma unroll
    for (int i = 0; i < 4; ++i)
#pragma unroll
      for (int j = 0; j < 2; ++j) acc[i][j] = wmma_bf16(af[i], bw[j], acc[i][j]);
    __syncthreads();                    // all reads of 'cur' done before it is reused
  }

  int lcol = lane & 15, lrow8 = (lane >> 4) << 3;
#pragma unroll
  for (int i = 0; i < 4; ++i)
#pragma unroll
    for (int j = 0; j < 2; ++j) {
      int row0 = bm + wm + i * 16 + lrow8;
      int col  = bn + wn + j * 16 + lcol;
#pragma unroll
      for (int rr = 0; rr < 8; ++rr) {
        int row = row0 + rr;
        size_t idx = (size_t)row * Ndim + col;
        float c = acc[i][j][rr];
        if (EPI == EPI_BF16) {
          outB[idx] = (bf16)c;
        } else if (EPI == EPI_F32) {
          outF[idx] = c;
        } else if (EPI == EPI_SILU) {
          outB[idx] = (bf16)(c / (1.f + __expf(-c)));
        } else if (EPI == EPI_ATTN) {
          float g = gpre[idx];
          outF[idx] = resid[idx] + c / (1.f + __expf(-g));
        } else {
          outF[idx] += coef[(size_t)row * cs] * c;
        }
      }
    }
}

// ---------------- flash attention (Br=64 queries, Bc=64 keys, HD=64) ----------------
__global__ __launch_bounds__(256)
void k_flash_attn(const bf16* __restrict__ Q, const bf16* __restrict__ Km,
                  const bf16* __restrict__ V, bf16* __restrict__ O) {
  __shared__ __align__(16) bf16  sQ[64 * 64];
  __shared__ __align__(16) bf16  sK[64 * 64];
  __shared__ __align__(16) bf16  sVt[64 * 64];   // transposed: [hd][key]
  __shared__ __align__(16) float sS[64 * 64];
  __shared__ __align__(16) bf16  sP[64 * 64];
  __shared__ __align__(16) float sO[64 * 64];

  int t = threadIdx.x, lane = t & 31, wave = t >> 5;
  int qt = blockIdx.x, h = blockIdx.y, b = blockIdx.z;
  size_t base = (size_t)b * NN * DD + (size_t)h * HD;
  int wm = (wave >> 1) * 16;   // 4 waves in M
  int wn = (wave & 1) * 32;    // 2 waves in N
  int lcol = lane & 15, lrow8 = (lane >> 4) << 3;
  int r = t >> 2, c8 = (t & 3) * 16;

  { // load Q tile (async)
    const bf16* gq = Q + base + (size_t)(qt * 64 + r) * DD + c8;
    async_cp16(gq,     sQ + r * 64 + c8);
    async_cp16(gq + 8, sQ + r * 64 + c8 + 8);
  }
  for (int i = t; i < 64 * 64; i += 256) sO[i] = 0.f;
  float mrun = -3.0e38f, lrun = 0.f;   // valid for t < 64 (one row per thread)

  for (int kt = 0; kt < NN / 64; ++kt) {
    __syncthreads();                   // protect sK/sVt from prior readers
    { // K tile async row-major; V tile transposed through VGPRs
      const bf16* gk = Km + base + (size_t)(kt * 64 + r) * DD + c8;
      async_cp16(gk,     sK + r * 64 + c8);
      async_cp16(gk + 8, sK + r * 64 + c8 + 8);
      const bf16* gv = V + base + (size_t)(kt * 64 + r) * DD + c8;
      bf16 tmp[16];
      *(uint4*)tmp       = *(const uint4*)gv;
      *(uint4*)(tmp + 8) = *(const uint4*)(gv + 8);
#pragma unroll
      for (int j = 0; j < 16; ++j) sVt[(c8 + j) * 64 + r] = tmp[j];
    }
    wait_async<0>();
    __syncthreads();

    // S = (Q K^T) * scale   (wave tile 16x32 -> 2 frags)
    v8f sacc[2] = {};
#pragma unroll
    for (int kb = 0; kb < 64; kb += 32) {
      v16bf aq  = load_frag(sQ, wm, kb, 64, lane);
      v16bf bk0 = load_frag(sK, wn,      kb, 64, lane);
      v16bf bk1 = load_frag(sK, wn + 16, kb, 64, lane);
      sacc[0] = wmma_bf16(aq, bk0, sacc[0]);
      sacc[1] = wmma_bf16(aq, bk1, sacc[1]);
    }
#pragma unroll
    for (int j = 0; j < 2; ++j)
#pragma unroll
      for (int rr = 0; rr < 8; ++rr)
        sS[(wm + lrow8 + rr) * 64 + wn + j * 16 + lcol] = sacc[j][rr] * 0.125f;
    __syncthreads();

    // online softmax: thread t (<64) owns query row t
    if (t < 64) {
      float rmax = -3.0e38f;
      for (int j = 0; j < 64; ++j) rmax = fmaxf(rmax, sS[t * 64 + j]);
      float mnew = fmaxf(mrun, rmax);
      float scl = __expf(mrun - mnew);
      float sum = 0.f;
      for (int j = 0; j < 64; ++j) {
        float p = __expf(sS[t * 64 + j] - mnew);
        sP[t * 64 + j] = (bf16)p;
        sum += p;
      }
      lrun = lrun * scl + sum;
      mrun = mnew;
      for (int j = 0; j < 64; ++j) sO[t * 64 + j] *= scl;
    }
    __syncthreads();

    // O += P * V
    v8f oacc[2] = {};
#pragma unroll
    for (int kb = 0; kb < 64; kb += 32) {
      v16bf ap  = load_frag(sP,  wm,      kb, 64, lane);
      v16bf bv0 = load_frag(sVt, wn,      kb, 64, lane);
      v16bf bv1 = load_frag(sVt, wn + 16, kb, 64, lane);
      oacc[0] = wmma_bf16(ap, bv0, oacc[0]);
      oacc[1] = wmma_bf16(ap, bv1, oacc[1]);
    }
#pragma unroll
    for (int j = 0; j < 2; ++j)
#pragma unroll
      for (int rr = 0; rr < 8; ++rr)
        sO[(wm + lrow8 + rr) * 64 + wn + j * 16 + lcol] += oacc[j][rr];
  }
  __syncthreads();
  if (t < 64) {
    float invl = 1.f / lrun;
    bf16* go = O + base + (size_t)(qt * 64 + t) * DD;
    for (int j = 0; j < 64; ++j) go[j] = (bf16)(sO[t * 64 + j] * invl);
  }
}

// ---------------- MoE router: logits -> top-2 softmax coef[M,E] ----------------
__global__ __launch_bounds__(256)
void k_moe_gate(const bf16* __restrict__ xn, const float* __restrict__ wg,
                float* __restrict__ coef) {
  int t = threadIdx.x, lane = t & 31, g = t >> 5;
  int m = blockIdx.x * 8 + g;
  float acc[EE] = {};
  for (int d = lane; d < DD; d += 32) {
    float xv = (float)xn[(size_t)m * DD + d];
#pragma unroll
    for (int e = 0; e < EE; ++e) acc[e] += xv * wg[e * DD + d];
  }
#pragma unroll
  for (int e = 0; e < EE; ++e)
    for (int o = 16; o; o >>= 1) acc[e] += __shfl_down(acc[e], o, 32);
  if (lane == 0) {
    int i0 = 0; float v0 = acc[0];
#pragma unroll
    for (int e = 1; e < EE; ++e) if (acc[e] > v0) { v0 = acc[e]; i0 = e; }
    int i1 = -1; float v1 = -3.0e38f;
#pragma unroll
    for (int e = 0; e < EE; ++e) if (e != i0 && acc[e] > v1) { v1 = acc[e]; i1 = e; }
    float e1 = __expf(v1 - v0);
    float inv = 1.f / (1.f + e1);
#pragma unroll
    for (int e = 0; e < EE; ++e)
      coef[(size_t)m * EE + e] = (e == i0) ? inv : (e == i1 ? e1 * inv : 0.f);
  }
}

// ---------------- host orchestration ----------------
extern "C" void kernel_launch(void* const* d_in, const int* in_sizes, int n_in,
                              void* d_out, int out_size, void* d_ws, size_t ws_size,
                              hipStream_t stream) {
  (void)in_sizes; (void)n_in; (void)out_size; (void)ws_size;
  const float* x     = (const float*)d_in[0];
  const float* n1w   = (const float*)d_in[1];
  const float* n2w   = (const float*)d_in[2];
  const float* wq    = (const float*)d_in[3];
  const float* wk    = (const float*)d_in[4];
  const float* wv    = (const float*)d_in[5];
  const float* wo    = (const float*)d_in[6];
  const float* wg    = (const float*)d_in[7];
  const float* wgate = (const float*)d_in[8];
  const float* w1    = (const float*)d_in[9];
  const float* w2    = (const float*)d_in[10];
  float* out = (float*)d_out;

  char* ws = (char*)d_ws; size_t off = 0;
  auto give = [&](size_t bytes) -> void* {
    void* p = ws + off; off += (bytes + 255) & ~(size_t)255; return p;
  };
  bf16*  xn1   = (bf16*) give((size_t)MT * DD * 2);
  bf16*  qb    = (bf16*) give((size_t)MT * DD * 2);
  bf16*  kb    = (bf16*) give((size_t)MT * DD * 2);
  bf16*  vb    = (bf16*) give((size_t)MT * DD * 2);
  float* gpre  = (float*)give((size_t)MT * DD * 4);
  bf16*  attnb = (bf16*) give((size_t)MT * DD * 2);
  bf16*  xn2   = (bf16*) give((size_t)MT * DD * 2);
  float* coef  = (float*)give((size_t)MT * EE * 4);
  bf16*  hbuf  = (bf16*) give((size_t)MT * FF * 2);
  bf16*  wq_b  = (bf16*) give((size_t)DD * DD * 2);
  bf16*  wk_b  = (bf16*) give((size_t)DD * DD * 2);
  bf16*  wv_b  = (bf16*) give((size_t)DD * DD * 2);
  bf16*  wo_b  = (bf16*) give((size_t)DD * DD * 2);
  bf16*  wg_b  = (bf16*) give((size_t)DD * DD * 2);
  bf16*  w1_b  = (bf16*) give((size_t)EE * FF * DD * 2);
  bf16*  w2_b  = (bf16*) give((size_t)EE * DD * FF * 2);

  // weights fp32 -> bf16 (entire bf16 weight set ~140MB: L2 resident afterwards)
  k_cvt_bf16<<<2048, 256, 0, stream>>>(wq, wq_b, (long)DD * DD);
  k_cvt_bf16<<<2048, 256, 0, stream>>>(wk, wk_b, (long)DD * DD);
  k_cvt_bf16<<<2048, 256, 0, stream>>>(wv, wv_b, (long)DD * DD);
  k_cvt_bf16<<<2048, 256, 0, stream>>>(wo, wo_b, (long)DD * DD);
  k_cvt_bf16<<<2048, 256, 0, stream>>>(wg, wg_b, (long)DD * DD);
  k_cvt_bf16<<<4096, 256, 0, stream>>>(w1, w1_b, (long)EE * FF * DD);
  k_cvt_bf16<<<4096, 256, 0, stream>>>(w2, w2_b, (long)EE * DD * FF);

  // ---- attention block ----
  k_rmsnorm<<<MT, 256, 0, stream>>>(x, n1w, xn1);
  dim3 gP(DD / 128, MT / 128);
  k_gemm<EPI_BF16><<<gP, 256, 0, stream>>>(xn1, wq_b, DD, DD, nullptr, qb, nullptr, nullptr, nullptr, 0);
  k_gemm<EPI_BF16><<<gP, 256, 0, stream>>>(xn1, wk_b, DD, DD, nullptr, kb, nullptr, nullptr, nullptr, 0);
  k_gemm<EPI_BF16><<<gP, 256, 0, stream>>>(xn1, wv_b, DD, DD, nullptr, vb, nullptr, nullptr, nullptr, 0);
  k_gemm<EPI_F32><<<gP, 256, 0, stream>>>(xn1, wg_b, DD, DD, gpre, nullptr, nullptr, nullptr, nullptr, 0);

  k_flash_attn<<<dim3(NN / 64, HH, BB), 256, 0, stream>>>(qb, kb, vb, attnb);

  // out = x + sigmoid(gpre) * (attn @ wo^T)
  k_gemm<EPI_ATTN><<<gP, 256, 0, stream>>>(attnb, wo_b, DD, DD, out, nullptr, gpre, x, nullptr, 0);

  // ---- MoE block ----
  k_rmsnorm<<<MT, 256, 0, stream>>>(out, n2w, xn2);
  k_moe_gate<<<MT / 8, 256, 0, stream>>>(xn2, wgate, coef);
  dim3 gUp(FF / 128, MT / 128), gDn(DD / 128, MT / 128);
  for (int e = 0; e < EE; ++e) {
    k_gemm<EPI_SILU><<<gUp, 256, 0, stream>>>(xn2, w1_b + (size_t)e * FF * DD,
                                              FF, DD, nullptr, hbuf, nullptr, nullptr, nullptr, 0);
    k_gemm<EPI_ACC><<<gDn, 256, 0, stream>>>(hbuf, w2_b + (size_t)e * DD * FF,
                                             DD, FF, out, nullptr, nullptr, nullptr, coef + e, EE);
  }
}
